// GOMemoryBlock_4063039062433
// MI455X (gfx1250) — matrix operand order repaired and checked by reading, not compiled, not run on hardware
//
#include <hip/hip_runtime.h>
#include <hip/hip_bf16.h>

#define T_GO   8000
#define DIN    768
#define DM     1280
#define NHEAD  8
#define DHEAD  160
#define HID    256
#define BATCH  16
#define NODES  256
#define NEDGE  48000
#define WPR    250      // 8000 bits / 32 per row

typedef __bf16 bf16_t;
typedef __attribute__((ext_vector_type(16))) __bf16 v16bf;
typedef __attribute__((ext_vector_type(8)))  __bf16 v8bf;
typedef __attribute__((ext_vector_type(4)))  __bf16 v4bf;
typedef __attribute__((ext_vector_type(8)))  float   v8f;
typedef __attribute__((ext_vector_type(4)))  float   v4f;

union V16U { v16bf v; v8bf h[2]; };

#define BM 128
#define BN 128
#define BK 32
#define BKP (BK + 8)

#define FLAG_RELU  1
#define FLAG_RESID 2
#define FLAG_OUTBF 4   // store bf16 TRANSPOSED: Cb[n*M + m]  (for weight composition)

// ---------------------------------------------------------------------------
// Tiled GEMM: C[M,N] = act(A[M,K](f32) * Bt[N,K](bf16, pre-transposed) + bias
//                         (+ resid)).
// 256 threads = 8 waves; wave tile 32x64 = 2x4 WMMA 16x16 f32 accumulators.
// LDS double-buffered; vectorized b128 staging on both operands.
// ---------------------------------------------------------------------------
__global__ __launch_bounds__(256) void gemm_bf16_kernel(
    const float* __restrict__ A, int lda,
    const bf16_t* __restrict__ Bt,          // [N][K], k contiguous
    const float* __restrict__ bias,
    const float* __restrict__ resid,
    float* __restrict__ Cf, bf16_t* __restrict__ Cb,
    int M, int N, int K, int flags)
{
  __shared__ __align__(16) bf16_t As[2][BM][BKP];
  __shared__ __align__(16) bf16_t Bs[2][BN][BKP];

  const int tid  = threadIdx.x;
  const int lane = tid & 31;
  const int wid  = tid >> 5;
  const int wm   = wid & 3;        // 4 waves along M (4*32 = 128)
  const int wn   = wid >> 2;       // 2 waves along N (2*64 = 128)
  const int l16  = lane & 15;
  const int hi   = lane >> 4;
  const int m0   = blockIdx.y * BM;
  const int n0   = blockIdx.x * BN;

  v8f acc[2][4] = {};

  // --- staging helpers (vectorized) ---
  auto stageA = [&](int k0, int buf) {
#pragma unroll
    for (int it = 0; it < (BM * BK) / (256 * 4); ++it) {      // 4 iterations
      int e = (tid + it * 256) * 4;
      int m = e / BK, k = e % BK;
      int gm = m0 + m;
      v4f v = {};
      if (gm < M) v = *(const v4f*)(A + (size_t)gm * lda + (k0 + k));
      v4bf o;
#pragma unroll
      for (int j = 0; j < 4; ++j) o[j] = (bf16_t)v[j];
      *(v4bf*)&As[buf][m][k] = o;
    }
  };
  auto stageB = [&](int k0, int buf) {
#pragma unroll
    for (int it = 0; it < (BN * BK) / (256 * 8); ++it) {      // 2 iterations
      int e = (tid + it * 256) * 8;
      int n = e / BK, k = e % BK;
      *(v8bf*)&Bs[buf][n][k] =
          *(const v8bf*)(Bt + (size_t)(n0 + n) * K + (k0 + k));
    }
  };

  const int NT = K / BK;
  stageA(0, 0);
  stageB(0, 0);
  __syncthreads();

  for (int kt = 0; kt < NT; ++kt) {
    const int cur = kt & 1;
    if (kt + 1 < NT) {                  // prefetch next tile into other buffer
      stageA((kt + 1) * BK, cur ^ 1);
      stageB((kt + 1) * BK, cur ^ 1);
    }

    V16U a[2], b[4];
#pragma unroll
    for (int mi = 0; mi < 2; ++mi) {
      int m  = wm * 32 + mi * 16 + l16;
      int kb = hi * 8;                  // A frag: lanes 0-15 K0-7/16-23, 16-31 K8-15/24-31
      a[mi].h[0] = *(const v8bf*)&As[cur][m][kb];
      a[mi].h[1] = *(const v8bf*)&As[cur][m][kb + 16];
    }
#pragma unroll
    for (int ni = 0; ni < 4; ++ni) {
      int n  = wn * 64 + ni * 16 + l16;
      int kb = hi * 8;                  // B frag: lanes 0-15 K0-7/16-23, 16-31 K8-15/24-31
      b[ni].h[0] = *(const v8bf*)&Bs[cur][n][kb];
      b[ni].h[1] = *(const v8bf*)&Bs[cur][n][kb + 16];
    }
#pragma unroll
    for (int mi = 0; mi < 2; ++mi)
#pragma unroll
      for (int ni = 0; ni < 4; ++ni)
        acc[mi][ni] = __builtin_amdgcn_wmma_f32_16x16x32_bf16(
            false, a[mi].v, false, b[ni].v, (short)0, acc[mi][ni], false, false);
    __syncthreads();
  }

#pragma unroll
  for (int mi = 0; mi < 2; ++mi) {
#pragma unroll
    for (int ni = 0; ni < 4; ++ni) {
      int n_g = n0 + wn * 64 + ni * 16 + l16;
      float bv = bias ? bias[n_g] : 0.0f;
#pragma unroll
      for (int i = 0; i < 8; ++i) {
        int m_g = m0 + wm * 32 + mi * 16 + hi * 8 + i;  // C/D: vgpr i -> M=i (+8 for hi lanes)
        if (m_g < M) {
          float v = acc[mi][ni][i] + bv;
          if (flags & FLAG_RESID) v += resid[(size_t)m_g * N + n_g];
          if (flags & FLAG_RELU)  v = fmaxf(v, 0.0f);
          if (flags & FLAG_OUTBF) Cb[(size_t)n_g * M + m_g] = (bf16_t)v;   // transposed
          else                    Cf[(size_t)m_g * N + n_g] = v;
        }
      }
    }
  }
}

// ---------------------------------------------------------------------------
// f32 [K,N] -> bf16 transposed [N,K], 32x32 LDS tiles (coalesced both sides)
// ---------------------------------------------------------------------------
__global__ __launch_bounds__(256) void transpose_cvt_kernel(
    const float* __restrict__ W, bf16_t* __restrict__ Wt, int K, int N)
{
  __shared__ float tile[32][33];
  int k0 = blockIdx.x * 32, n0 = blockIdx.y * 32;
  int tx = threadIdx.x & 31, ty = threadIdx.x >> 5;
  for (int r = ty; r < 32; r += 8)
    tile[r][tx] = W[(size_t)(k0 + r) * N + (n0 + tx)];
  __syncthreads();
  for (int r = ty; r < 32; r += 8)
    Wt[(size_t)(n0 + r) * K + (k0 + tx)] = (bf16_t)tile[tx][r];
}

// ---------------------------------------------------------------------------
// Adjacency bitmap + symmetric-normalized SpMM (exact dedup like .set(1.0))
// ---------------------------------------------------------------------------
__global__ void zero_u32_kernel(unsigned* p, int n) {
  int i = blockIdx.x * 256 + threadIdx.x;
  if (i < n) p[i] = 0u;
}
__global__ void diag_bits_kernel(unsigned* bm) {
  int i = blockIdx.x * 256 + threadIdx.x;
  if (i < T_GO) atomicOr(&bm[(size_t)i * WPR + (i >> 5)], 1u << (i & 31));
}
__global__ void edge_bits_kernel(const int* __restrict__ ei, unsigned* bm) {
  int e = blockIdx.x * 256 + threadIdx.x;
  if (e < NEDGE) {
    int r = ei[e], c = ei[NEDGE + e];
    atomicOr(&bm[(size_t)r * WPR + (c >> 5)], 1u << (c & 31));
  }
}
__global__ void degree_kernel(const unsigned* __restrict__ bm, float* __restrict__ dinv) {
  int i = blockIdx.x * 256 + threadIdx.x;
  if (i < T_GO) {
    const unsigned* row = bm + (size_t)i * WPR;
    int c = 0;
    for (int w = 0; w < WPR; ++w) c += __popc(row[w]);
    dinv[i] = rsqrtf(fmaxf((float)c, 1e-12f));
  }
}
// out[i,:] = dinv[i] * sum_{j in row} dinv[j] * X[j,:]  (+ resid)
__global__ __launch_bounds__(256) void spmm_kernel(
    const unsigned* __restrict__ bm, const float* __restrict__ dinv,
    const float* __restrict__ X, const float* __restrict__ resid,
    float* __restrict__ out)
{
  __shared__ unsigned wrow[WPR];
  int row = blockIdx.x, tid = threadIdx.x;
  for (int w = tid; w < WPR; w += 256) wrow[w] = bm[(size_t)row * WPR + w];
  __syncthreads();
  float acc[5] = {0, 0, 0, 0, 0};
  for (int w = 0; w < WPR; ++w) {
    unsigned bits = wrow[w];
    while (bits) {
      int j = (w << 5) + __builtin_ctz(bits);
      bits &= bits - 1;
      float dj = dinv[j];
      const float* xp = X + (size_t)j * DM;
#pragma unroll
      for (int c = 0; c < 5; ++c) acc[c] += dj * xp[tid + c * 256];
    }
  }
  float di = dinv[row];
#pragma unroll
  for (int c = 0; c < 5; ++c) {
    int col = tid + c * 256;
    float v = di * acc[c];
    if (resid) v += resid[(size_t)row * DM + col];
    out[(size_t)row * DM + col] = v;
  }
}

__global__ void goz_kernel(const float* __restrict__ hisa, const float* __restrict__ hpart,
                           const float* __restrict__ alpha_logit, float* __restrict__ out)
{
  size_t i = (size_t)blockIdx.x * 256 + threadIdx.x;
  if (i < (size_t)T_GO * DM) {
    int row = (int)(i / DM);
    float a = 1.0f / (1.0f + expf(-alpha_logit[row]));
    out[i] = a * hisa[i] + (1.0f - a) * hpart[i];
  }
}

// out[n] = sum_k bsrc[k]*W[k,n] + badd[n]   (bias folding for composed weights)
__global__ void bias_compose_kernel(const float* __restrict__ bsrc, const float* __restrict__ W,
                                    const float* __restrict__ badd, float* __restrict__ out,
                                    int K, int N)
{
  int n = blockIdx.x * 256 + threadIdx.x;
  if (n < N) {
    float s = badd[n];
    for (int k = 0; k < K; ++k) s += bsrc[k] * W[(size_t)k * N + n];
    out[n] = s;
  }
}

// ---------------------------------------------------------------------------
// Attention 1: B=16 queries over shared T=8000 memory (per head)
// ---------------------------------------------------------------------------
__global__ __launch_bounds__(256) void attn1_scores_kernel(
    const float* __restrict__ qh, const float* __restrict__ Kh, float* __restrict__ S)
{
  __shared__ __align__(16) float q_s[BATCH][DHEAD];
  int h = blockIdx.x;
  int c0 = blockIdx.y * 1000;
  int tid = threadIdx.x;
  for (int e = tid; e < BATCH * DHEAD; e += 256) {
    int b = e / DHEAD, d = e % DHEAD;
    q_s[b][d] = qh[(size_t)b * DM + h * DHEAD + d];
  }
  __syncthreads();
  const float scale = 0.07905694150420949f;   // 1/sqrt(160)
  for (int t = c0 + tid; t < c0 + 1000 && t < T_GO; t += 256) {
    const float* kp = Kh + (size_t)t * DM + h * DHEAD;
    float acc[BATCH];
#pragma unroll
    for (int b = 0; b < BATCH; ++b) acc[b] = 0.0f;
    for (int d = 0; d < DHEAD; d += 4) {
      v4f kv = *(const v4f*)(kp + d);
#pragma unroll
      for (int b = 0; b < BATCH; ++b) {
        v4f qv = *(const v4f*)&q_s[b][d];
        acc[b] += qv[0] * kv[0] + qv[1] * kv[1] + qv[2] * kv[2] + qv[3] * kv[3];
      }
    }
    for (int b = 0; b < BATCH; ++b)
      S[((size_t)b * NHEAD + h) * T_GO + t] = acc[b] * scale;
  }
}

__global__ __launch_bounds__(256) void softmax_rows_kernel(float* __restrict__ S, int ncols)
{
  __shared__ float red[256];
  int row = blockIdx.x, tid = threadIdx.x;
  float* p = S + (size_t)row * ncols;
  float m = -1e30f;
  for (int c = tid; c < ncols; c += 256) m = fmaxf(m, p[c]);
  red[tid] = m; __syncthreads();
  for (int o = 128; o > 0; o >>= 1) { if (tid < o) red[tid] = fmaxf(red[tid], red[tid + o]); __syncthreads(); }
  m = red[0]; __syncthreads();
  float s = 0.0f;
  for (int c = tid; c < ncols; c += 256) { float e = expf(p[c] - m); p[c] = e; s += e; }
  red[tid] = s; __syncthreads();
  for (int o = 128; o > 0; o >>= 1) { if (tid < o) red[tid] += red[tid + o]; __syncthreads(); }
  float inv = 1.0f / red[0];
  for (int c = tid; c < ncols; c += 256) p[c] *= inv;
}

__global__ __launch_bounds__(160) void attn1_ctx_kernel(
    const float* __restrict__ S, const float* __restrict__ Vh, float* __restrict__ ctx)
{
  __shared__ float p_s[BATCH][64];
  int h = blockIdx.x, tid = threadIdx.x;
  float acc[BATCH];
#pragma unroll
  for (int b = 0; b < BATCH; ++b) acc[b] = 0.0f;
  for (int t0 = 0; t0 < T_GO; t0 += 64) {
    for (int e = tid; e < BATCH * 64; e += 160) {
      int b = e >> 6, tt = e & 63;
      p_s[b][tt] = S[((size_t)b * NHEAD + h) * T_GO + t0 + tt];
    }
    __syncthreads();
    for (int tt = 0; tt < 64; ++tt) {
      float v = Vh[(size_t)(t0 + tt) * DM + h * DHEAD + tid];
#pragma unroll
      for (int b = 0; b < BATCH; ++b) acc[b] += p_s[b][tt] * v;
    }
    __syncthreads();
  }
  for (int b = 0; b < BATCH; ++b)
    ctx[(size_t)b * DM + h * DHEAD + tid] = acc[b];
}

// ---------------------------------------------------------------------------
// Attention 2: per (b,h) over 256 structure nodes (batch_vec is
// repeat(arange(B), NODES) so to_dense_batch == reshape, mask all-true)
// ---------------------------------------------------------------------------
__global__ __launch_bounds__(256) void attn2_kernel(
    const float* __restrict__ q2, const float* __restrict__ k2,
    const float* __restrict__ v2, float* __restrict__ ctx2)
{
  __shared__ float q_s[DHEAD];
  __shared__ float p_s[NODES];
  __shared__ float red[256];
  int h = blockIdx.x, b = blockIdx.y, tid = threadIdx.x;
  if (tid < DHEAD) q_s[tid] = q2[(size_t)b * DM + h * DHEAD + tid];
  __syncthreads();
  const float scale = 0.07905694150420949f;
  const float* kp = k2 + ((size_t)(b * NODES + tid)) * DM + h * DHEAD;
  float dot = 0.0f;
  for (int d = 0; d < DHEAD; ++d) dot += q_s[d] * kp[d];
  dot *= scale;
  red[tid] = dot; __syncthreads();
  for (int o = 128; o > 0; o >>= 1) { if (tid < o) red[tid] = fmaxf(red[tid], red[tid + o]); __syncthreads(); }
  float m = red[0]; __syncthreads();
  float e = expf(dot - m);
  red[tid] = e; __syncthreads();
  for (int o = 128; o > 0; o >>= 1) { if (tid < o) red[tid] += red[tid + o]; __syncthreads(); }
  float inv = 1.0f / red[0];
  p_s[tid] = e * inv;
  __syncthreads();
  if (tid < DHEAD) {
    float acc = 0.0f;
    for (int n = 0; n < NODES; ++n)
      acc += p_s[n] * v2[((size_t)(b * NODES + n)) * DM + h * DHEAD + tid];
    ctx2[(size_t)b * DM + h * DHEAD + tid] = acc;
  }
}

__global__ __launch_bounds__(256) void layernorm_kernel(
    const float* __restrict__ x, const float* __restrict__ g,
    const float* __restrict__ b, float* __restrict__ out, int ncols)
{
  __shared__ float red[256];
  int row = blockIdx.x, tid = threadIdx.x;
  const float* xp = x + (size_t)row * ncols;
  float s = 0.0f;
  for (int c = tid; c < ncols; c += 256) s += xp[c];
  red[tid] = s; __syncthreads();
  for (int o = 128; o > 0; o >>= 1) { if (tid < o) red[tid] += red[tid + o]; __syncthreads(); }
  float mean = red[0] / ncols; __syncthreads();
  float v = 0.0f;
  for (int c = tid; c < ncols; c += 256) { float d = xp[c] - mean; v += d * d; }
  red[tid] = v; __syncthreads();
  for (int o = 128; o > 0; o >>= 1) { if (tid < o) red[tid] += red[tid + o]; __syncthreads(); }
  float inv = rsqrtf(red[0] / ncols + 1e-5f);
  for (int c = tid; c < ncols; c += 256)
    out[(size_t)row * ncols + c] = (xp[c] - mean) * inv * g[c] + b[c];
}

// ---------------------------------------------------------------------------
extern "C" void kernel_launch(void* const* d_in, const int* in_sizes, int n_in,
                              void* d_out, int out_size, void* d_ws, size_t ws_size,
                              hipStream_t stream)
{
  (void)in_sizes; (void)n_in; (void)out_size; (void)ws_size;
  // inputs (x_dense[0], mask_tok[1], batch_vec[4] are unused by the reference output)
  const float* esm_pooled = (const float*)d_in[2];
  const float* h_node     = (const float*)d_in[3];
  const float* biobert    = (const float*)d_in[5];
  const int*   edge_is_a  = (const int*)d_in[6];
  const int*   edge_part  = (const int*)d_in[7];
  const float* lin1_w = (const float*)d_in[8];  const float* lin1_b = (const float*)d_in[9];
  const float* lin2_w = (const float*)d_in[10]; const float* lin2_b = (const float*)d_in[11];
  const float* part_w = (const float*)d_in[12]; const float* part_b = (const float*)d_in[13];
  const float* res_w  = (const float*)d_in[14]; const float* res_b  = (const float*)d_in[15];
  const float* alpha_logit = (const float*)d_in[16];
  const float* esmq_w = (const float*)d_in[17]; const float* esmq_b = (const float*)d_in[18];
  const float* k_w    = (const float*)d_in[19]; const float* k_b    = (const float*)d_in[20];
  const float* v_w    = (const float*)d_in[21]; const float* v_b    = (const float*)d_in[22];
  const float* a1_wq  = (const float*)d_in[23]; const float* a1_bq  = (const float*)d_in[24];
  const float* a1_wk  = (const float*)d_in[25]; const float* a1_bk  = (const float*)d_in[26];
  const float* a1_wv  = (const float*)d_in[27]; const float* a1_bv  = (const float*)d_in[28];
  const float* a1_wo  = (const float*)d_in[29]; const float* a1_bo  = (const float*)d_in[30];
  const float* ln1_g  = (const float*)d_in[31]; const float* ln1_b  = (const float*)d_in[32];
  const float* a2_wq  = (const float*)d_in[33]; const float* a2_bq  = (const float*)d_in[34];
  const float* a2_wk  = (const float*)d_in[35]; const float* a2_bk  = (const float*)d_in[36];
  const float* a2_wv  = (const float*)d_in[37]; const float* a2_bv  = (const float*)d_in[38];
  const float* a2_wo  = (const float*)d_in[39]; const float* a2_bo  = (const float*)d_in[40];
  const float* ln2_g  = (const float*)d_in[41]; const float* ln2_b  = (const float*)d_in[42];

  // ---- workspace carve-up ----
  char* ws = (char*)d_ws;
  size_t off = 0;
  auto alloc = [&](size_t bytes) -> char* {
    char* p = ws + off;
    off += (bytes + 255) & ~(size_t)255;
    return p;
  };
  const size_t bigB = (size_t)T_GO * DM * 4;
  float* A1 = (float*)alloc(bigB);
  float* A2 = (float*)alloc(bigB);
  float* A3 = (float*)alloc(bigB);
  float* A4 = (float*)alloc(bigB);
  // bf16 weights, ALL stored transposed [N][K]
  bf16_t* w_lin1 = (bf16_t*)alloc((size_t)DIN * DM * 2);
  bf16_t* w_lin2 = (bf16_t*)alloc((size_t)DM * DM * 2);
  bf16_t* w_res  = (bf16_t*)alloc((size_t)DIN * DM * 2);
  bf16_t* w_part = (bf16_t*)alloc((size_t)DM * DM * 2);
  bf16_t* w_a1wk = (bf16_t*)alloc((size_t)DM * DM * 2);
  bf16_t* w_a1wv = (bf16_t*)alloc((size_t)DM * DM * 2);
  bf16_t* w_a1wq = (bf16_t*)alloc((size_t)DM * DM * 2);
  bf16_t* w_a1wo = (bf16_t*)alloc((size_t)DM * DM * 2);
  bf16_t* w_a2wq = (bf16_t*)alloc((size_t)DM * DM * 2);
  bf16_t* w_a2wk = (bf16_t*)alloc((size_t)HID * DM * 2);
  bf16_t* w_a2wv = (bf16_t*)alloc((size_t)HID * DM * 2);
  bf16_t* w_a2wo = (bf16_t*)alloc((size_t)DM * DM * 2);
  bf16_t* w_Kp   = (bf16_t*)alloc((size_t)DM * DM * 2);   // (k_w @ a1_wk)^T
  bf16_t* w_Vp   = (bf16_t*)alloc((size_t)DM * DM * 2);   // (v_w @ a1_wv)^T
  bf16_t* w_Qp   = (bf16_t*)alloc((size_t)DM * DM * 2);   // (esmq_w @ a1_wq)^T
  float* bkp = (float*)alloc(DM * 4);
  float* bvp = (float*)alloc(DM * 4);
  float* bqp = (float*)alloc(DM * 4);
  // bitmaps alias later k2/v2 buffers (disjoint lifetimes)
  const size_t k2B = (size_t)BATCH * NODES * DM * 4;       // 21 MB > 8 MB bitmap
  char* reg1 = alloc(k2B);  unsigned* bm1 = (unsigned*)reg1;  float* k2buf = (float*)reg1;
  char* reg2 = alloc(k2B);  unsigned* bm2 = (unsigned*)reg2;  float* v2buf = (float*)reg2;
  float* dinv1 = (float*)alloc(T_GO * 4);
  float* dinv2 = (float*)alloc(T_GO * 4);
  float* Sbuf  = (float*)alloc((size_t)BATCH * NHEAD * T_GO * 4);
  const size_t smB = (size_t)BATCH * DM * 4;
  float* qh     = (float*)alloc(smB);
  float* ctx1   = (float*)alloc(smB);
  float* go_pre = (float*)alloc(smB);
  float* go_vec = (float*)alloc(smB);
  float* q2buf  = (float*)alloc(smB);
  float* ctx2   = (float*)alloc(smB);
  float* outpre = (float*)alloc(smB);

  auto cvtT = [&](const float* src, bf16_t* dst, int K, int N) {
    transpose_cvt_kernel<<<dim3(K / 32, N / 32), 256, 0, stream>>>(src, dst, K, N);
  };
  auto gemm = [&](const float* Ap, int lda, const bf16_t* Btp, int Np,
                  const float* bias, const float* resid,
                  float* Cf, bf16_t* Cb, int Mp, int Kp, int flags) {
    dim3 g(Np / BN, (Mp + BM - 1) / BM);
    gemm_bf16_kernel<<<g, 256, 0, stream>>>(Ap, lda, Btp, bias, resid, Cf, Cb,
                                            Mp, Np, Kp, flags);
  };

  // ---- weight conversion to transposed bf16 ----
  cvtT(lin1_w, w_lin1, DIN, DM);   cvtT(lin2_w, w_lin2, DM, DM);
  cvtT(res_w,  w_res,  DIN, DM);   cvtT(part_w, w_part, DM, DM);
  cvtT(a1_wk,  w_a1wk, DM, DM);    cvtT(a1_wv,  w_a1wv, DM, DM);
  cvtT(a1_wq,  w_a1wq, DM, DM);    cvtT(a1_wo,  w_a1wo, DM, DM);
  cvtT(a2_wq,  w_a2wq, DM, DM);    cvtT(a2_wk,  w_a2wk, HID, DM);
  cvtT(a2_wv,  w_a2wv, HID, DM);   cvtT(a2_wo,  w_a2wo, DM, DM);

  // ---- adjacency bitmaps, degrees ----
  int nwords = T_GO * WPR;
  zero_u32_kernel<<<(nwords + 255) / 256, 256, 0, stream>>>(bm1, nwords);
  zero_u32_kernel<<<(nwords + 255) / 256, 256, 0, stream>>>(bm2, nwords);
  diag_bits_kernel<<<(T_GO + 255) / 256, 256, 0, stream>>>(bm1);
  diag_bits_kernel<<<(T_GO + 255) / 256, 256, 0, stream>>>(bm2);
  edge_bits_kernel<<<(NEDGE + 255) / 256, 256, 0, stream>>>(edge_is_a, bm1);
  edge_bits_kernel<<<(NEDGE + 255) / 256, 256, 0, stream>>>(edge_part, bm2);
  degree_kernel<<<(T_GO + 255) / 256, 256, 0, stream>>>(bm1, dinv1);
  degree_kernel<<<(T_GO + 255) / 256, 256, 0, stream>>>(bm2, dinv2);

  // ---- dual GO encoder ----
  gemm(biobert, DIN, w_lin1, DM, lin1_b, nullptr, A1, nullptr, T_GO, DIN, FLAG_RELU); // base1
  gemm(A1, DM, w_lin2, DM, lin2_b, nullptr, A2, nullptr, T_GO, DM, 0);                // base
  gemm(biobert, DIN, w_res, DM, res_b, nullptr, A3, nullptr, T_GO, DIN, 0);           // r
  spmm_kernel<<<T_GO, 256, 0, stream>>>(bm1, dinv1, A2, A3, A1);                      // h_is_a
  spmm_kernel<<<T_GO, 256, 0, stream>>>(bm2, dinv2, A2, nullptr, A4);                 // Ahat_part@base
  gemm(A4, DM, w_part, DM, part_b, A3, A2, nullptr, T_GO, DM, FLAG_RESID);            // h_part
  {
    size_t n = (size_t)T_GO * DM;
    goz_kernel<<<(unsigned)((n + 255) / 256), 256, 0, stream>>>(A1, A2, alpha_logit, A4); // go_z
  }

  // ---- fold (k_w,a1_wk) / (v_w,a1_wv) / (esmq_w,a1_wq) into single weights ----
  gemm(k_w,    DM, w_a1wk, DM, nullptr, nullptr, nullptr, w_Kp, DM, DM, FLAG_OUTBF);
  gemm(v_w,    DM, w_a1wv, DM, nullptr, nullptr, nullptr, w_Vp, DM, DM, FLAG_OUTBF);
  gemm(esmq_w, DM, w_a1wq, DM, nullptr, nullptr, nullptr, w_Qp, DM, DM, FLAG_OUTBF);
  bias_compose_kernel<<<(DM + 255) / 256, 256, 0, stream>>>(k_b, a1_wk, a1_bk, bkp, DM, DM);
  bias_compose_kernel<<<(DM + 255) / 256, 256, 0, stream>>>(v_b, a1_wv, a1_bv, bvp, DM, DM);
  bias_compose_kernel<<<(DM + 255) / 256, 256, 0, stream>>>(esmq_b, a1_wq, a1_bq, bqp, DM, DM);

  gemm(A4, DM, w_Kp, DM, bkp, nullptr, A1, nullptr, T_GO, DM, 0);           // kh
  gemm(A4, DM, w_Vp, DM, bvp, nullptr, A2, nullptr, T_GO, DM, 0);           // vh
  gemm(esm_pooled, DM, w_Qp, DM, bqp, nullptr, qh, nullptr, BATCH, DM, 0);  // qh

  // ---- cross-attention 1 over GO memory ----
  attn1_scores_kernel<<<dim3(NHEAD, 8), 256, 0, stream>>>(qh, A1, Sbuf);
  softmax_rows_kernel<<<BATCH * NHEAD, 256, 0, stream>>>(Sbuf, T_GO);
  attn1_ctx_kernel<<<NHEAD, 160, 0, stream>>>(Sbuf, A2, ctx1);
  gemm(ctx1, DM, w_a1wo, DM, a1_bo, nullptr, go_pre, nullptr, BATCH, DM, 0);
  layernorm_kernel<<<BATCH, 256, 0, stream>>>(go_pre, ln1_g, ln1_b, go_vec, DM);

  // ---- go -> structure cross attention ----
  gemm(h_node, HID, w_a2wk, DM, a2_bk, nullptr, k2buf, nullptr, BATCH * NODES, HID, 0);
  gemm(h_node, HID, w_a2wv, DM, a2_bv, nullptr, v2buf, nullptr, BATCH * NODES, HID, 0);
  gemm(go_vec, DM, w_a2wq, DM, a2_bq, nullptr, q2buf, nullptr, BATCH, DM, 0);
  attn2_kernel<<<dim3(NHEAD, BATCH), 256, 0, stream>>>(q2buf, k2buf, v2buf, ctx2);
  gemm(ctx2, DM, w_a2wo, DM, a2_bo, go_vec, outpre, nullptr, BATCH, DM, FLAG_RESID);
  layernorm_kernel<<<BATCH, 256, 0, stream>>>(outpre, ln2_g, ln2_b, (float*)d_out, DM);
}
